// TPGNNModel_82660940579240
// MI455X (gfx1250) — matrix operand order repaired and checked by reading, not compiled, not run on hardware
//
#include <hip/hip_runtime.h>
#include <cstdint>

typedef __attribute__((ext_vector_type(16))) _Float16 v16h;
typedef __attribute__((ext_vector_type(8)))  float    v8f;

#define HDIM  128
#define HEADS 4
#define DHD   32
#define ODIM  64

// ---------------------------------------------------------------- fills
__global__ void fill_f32_kernel(float* p, int n, float v) {
  int i = blockIdx.x * blockDim.x + threadIdx.x;
  if (i < n) p[i] = v;
}
__global__ void fill_u32_kernel(unsigned* p, int n, unsigned v) {
  int i = blockIdx.x * blockDim.x + threadIdx.x;
  if (i < n) p[i] = v;
}
__global__ void f32_to_f16_kernel(const float* __restrict__ s, _Float16* __restrict__ d, int n) {
  int i = blockIdx.x * blockDim.x + threadIdx.x;
  if (i < n) d[i] = (_Float16)s[i];
}

// ------------------------------------------------ input column stats (4 cols)
__global__ void col_stats4_kernel(const float* __restrict__ x, float* sum4, float* sq4, int N) {
  int col = threadIdx.x & 3;                  // blockDim = 128
  int r    = blockIdx.x * 1024 + (threadIdx.x >> 2);
  int rend = blockIdx.x * 1024 + 1024; if (rend > N) rend = N;
  float s = 0.f, q = 0.f;
  for (; r < rend; r += 32) { float v = x[r * 6 + 2 + col]; s += v; q += v * v; }
  atomicAdd(&sum4[col], s); atomicAdd(&sq4[col], q);
}
__global__ void in_stats_fin_kernel(const float* sum4, const float* sq4,
                                    float* mean4, float* inv4, int N) {
  int c = threadIdx.x;
  if (c < 4) {
    float m   = sum4[c] / (float)N;
    float var = (sq4[c] - (float)N * m * m) / (float)(N - 1);   // ddof=1
    float sd  = sqrtf(fmaxf(var, 0.f));
    sd = fmaxf(sd, 1e-8f);
    mean4[c] = m; inv4[c] = 1.f / sd;
  }
}

// ------------------------------------------------ input projection (K=4) + ReLU
__global__ void in_proj_kernel(const float* __restrict__ x, const float* __restrict__ w,
                               const float* __restrict__ b, const float* __restrict__ mean4,
                               const float* __restrict__ inv4,
                               float* __restrict__ h, _Float16* __restrict__ h16, int N) {
  int i = blockIdx.x * blockDim.x + threadIdx.x;
  if (i >= N * HDIM) return;
  int n = i >> 7, c = i & 127;
  float acc = b[c];
#pragma unroll
  for (int k = 0; k < 4; ++k)
    acc += (x[n * 6 + 2 + k] - mean4[k]) * inv4[k] * w[c * 4 + k];
  float hv = fmaxf(acc, 0.f);
  h[i] = hv; h16[i] = (_Float16)hv;
}

// ------------------------------------------------ WMMA GEMM: D = A(N,128) @ W(ncols,128)^T
// One wave produces a 16x64 strip (4 column tiles): the A fragment is loaded once
// per 32-wide K step and reused by 4 v_wmma_f32_16x16x32_f16 ops (A traffic /4).
// Fragments use contiguous K order (k = 16h + s) for both A and B. Compile-time flags keep the epilogue branch-free.
template <bool HAS_BIAS, bool HAS_RESID, bool HAS_DOUT, bool HAS_D16>
__global__ void wmma_gemm_kernel(const _Float16* __restrict__ A,
                                 const _Float16* __restrict__ W,
                                 const float* __restrict__ bias,   // per output col
                                 const float* __restrict__ resid,  // N x ncols
                                 float* __restrict__ Dout,         // N x ncols
                                 _Float16* __restrict__ D16,       // N x ncols
                                 int nrows, int ncols) {
  const int lane    = threadIdx.x & 31;
  const int wave    = threadIdx.x >> 5;
  const int groupsN = ncols >> 6;                      // 64-wide column groups
  const int item    = blockIdx.x * 8 + wave;
  const int total   = (nrows >> 4) * groupsN;
  if (item >= total) return;                           // wave-uniform: EXEC stays all-ones
  const int tm = item / groupsN, tg = item % groupsN;

  const int arow = tm * 16 + (lane & 15);
  const _Float16* __restrict__ Arow = A + (size_t)arow * HDIM;
  const int ahi = (lane >= 16) ? 8 : 0;                // A dword (K-pair) offset, hi half-wave
  const int wrow0 = tg * 64 + (lane & 15);             // W row of column-tile 0 for this lane
  const _Float16* __restrict__ Wb = W + (size_t)wrow0 * HDIM + ((lane >= 16) ? 16 : 0);

  __builtin_prefetch(Arow, 0, 0);                      // global_prefetch_b8
  __builtin_prefetch(Wb, 0, 0);

  v8f acc0 = {}, acc1 = {}, acc2 = {}, acc3 = {};
#pragma unroll
  for (int kk = 0; kk < HDIM; kk += 32) {
    const uint32_t* ap = (const uint32_t*)(Arow + kk);
    union { v16h v; uint32_t u[8]; } af;
#pragma unroll
    for (int i = 0; i < 4; ++i) {                      // A: 16 consecutive K halves (K=16h+s)
      af.u[i]     = ap[ahi + i];
      af.u[4 + i] = ap[ahi + 4 + i];
    }
#pragma unroll
    for (int t = 0; t < 4; ++t) {
      const uint32_t* bp = (const uint32_t*)(Wb + (size_t)t * 16 * HDIM + kk);
      union { v16h v; uint32_t u[8]; } bf;
#pragma unroll
      for (int i = 0; i < 8; ++i) bf.u[i] = bp[i];     // B: 16 consecutive K halves
      v8f* acc = (t == 0) ? &acc0 : (t == 1) ? &acc1 : (t == 2) ? &acc2 : &acc3;
      *acc = __builtin_amdgcn_wmma_f32_16x16x32_f16(
          false, af.v, false, bf.v, (short)0, *acc, false, false);
    }
  }

  const int rowBase = tm * 16 + ((lane >= 16) ? 8 : 0);
#pragma unroll
  for (int t = 0; t < 4; ++t) {
    const v8f& acc = (t == 0) ? acc0 : (t == 1) ? acc1 : (t == 2) ? acc2 : acc3;
    const int colBase = tg * 64 + t * 16 + (lane & 15);
    const float bv = HAS_BIAS ? bias[colBase] : 0.f;
#pragma unroll
    for (int i = 0; i < 8; ++i) {
      size_t idx = (size_t)(rowBase + i) * ncols + colBase;
      float v = acc[i] + bv;
      if (HAS_RESID) v += resid[idx];
      if (HAS_DOUT)  Dout[idx] = v;
      if (HAS_D16)   D16[idx]  = (_Float16)v;
    }
  }
}

// ------------------------------------------------ attention logits per (node, head)
__global__ void att_scores_kernel(const float* __restrict__ xh,
                                  const float* __restrict__ asrc, const float* __restrict__ adst,
                                  float* __restrict__ es, float* __restrict__ ed, int N) {
  int i = blockIdx.x * blockDim.x + threadIdx.x;
  if (i >= N * HEADS) return;
  int n = i >> 2, hd = i & 3;
  const float* xr = xh + (size_t)n * HDIM + hd * DHD;
  const float* as = asrc + hd * DHD;
  const float* ad = adst + hd * DHD;
  float s = 0.f, d = 0.f;
#pragma unroll
  for (int k = 0; k < DHD; ++k) { float v = xr[k]; s += v * as[k]; d += v * ad[k]; }
  es[i] = s; ed[i] = d;
}

// order-preserving float<->uint mapping for atomic max
__device__ __forceinline__ unsigned fkey(float f) {
  unsigned b = __float_as_uint(f);
  return b ^ ((unsigned)((int)b >> 31) | 0x80000000u);
}
__device__ __forceinline__ float funkey(unsigned u) {
  unsigned m = (u & 0x80000000u) ? 0x80000000u : 0xFFFFFFFFu;
  return __uint_as_float(u ^ m);
}
__device__ __forceinline__ void edge_sd(const int* __restrict__ ei, int E, int e, int& s, int& d) {
  if (e < E) { s = ei[e]; d = ei[E + e]; } else { s = e - E; d = e - E; }  // self-loops appended
}
__device__ __forceinline__ float leaky02(float v) { return v < 0.f ? 0.2f * v : v; }

__global__ void edge_max_kernel(const int* __restrict__ ei, int E, int ET,
                                const float* __restrict__ es, const float* __restrict__ ed,
                                unsigned* __restrict__ mkey) {
  int i = blockIdx.x * blockDim.x + threadIdx.x;
  if (i >= ET * HEADS) return;
  int e = i >> 2, hd = i & 3, s, d;
  edge_sd(ei, E, e, s, d);
  float v = leaky02(es[s * HEADS + hd] + ed[d * HEADS + hd]);
  atomicMax(&mkey[d * HEADS + hd], fkey(v));
}
__global__ void edge_sum_kernel(const int* __restrict__ ei, int E, int ET,
                                const float* __restrict__ es, const float* __restrict__ ed,
                                const unsigned* __restrict__ mkey, float* __restrict__ den) {
  int i = blockIdx.x * blockDim.x + threadIdx.x;
  if (i >= ET * HEADS) return;
  int e = i >> 2, hd = i & 3, s, d;
  edge_sd(ei, E, e, s, d);
  float v = leaky02(es[s * HEADS + hd] + ed[d * HEADS + hd]);
  float m = funkey(mkey[d * HEADS + hd]);
  atomicAdd(&den[d * HEADS + hd], __expf(v - m));
}
// one 32-lane wave per edge: lane covers 4 channels inside head = lane/8
__global__ void edge_agg_kernel(const int* __restrict__ ei, int E, int ET,
                                const float* __restrict__ es, const float* __restrict__ ed,
                                const unsigned* __restrict__ mkey, const float* __restrict__ den,
                                const float* __restrict__ xh, float* __restrict__ hn) {
  int i = blockIdx.x * blockDim.x + threadIdx.x;
  int e = i >> 5;
  if (e >= ET) return;
  int lane = i & 31, hd = lane >> 3, c0 = lane * 4, s, d;
  edge_sd(ei, E, e, s, d);
  float v = leaky02(es[s * HEADS + hd] + ed[d * HEADS + hd]);
  float m = funkey(mkey[d * HEADS + hd]);
  float alpha = __expf(v - m) / den[d * HEADS + hd];
  float4 xv = *(const float4*)(xh + (size_t)s * HDIM + c0);
  float* o = hn + (size_t)d * HDIM + c0;
  atomicAdd(o + 0, alpha * xv.x);
  atomicAdd(o + 1, alpha * xv.y);
  atomicAdd(o + 2, alpha * xv.z);
  atomicAdd(o + 3, alpha * xv.w);
}

// ------------------------------------------------ batchnorm (per-channel bias cancels)
__global__ void bn_stats_kernel(const float* __restrict__ hn, float* sum, float* sq, int N) {
  int c  = threadIdx.x;                           // blockDim = 128
  int r0 = blockIdx.x * 256;
  int r1 = r0 + 256; if (r1 > N) r1 = N;
  float s = 0.f, q = 0.f;
  for (int r = r0; r < r1; ++r) { float v = hn[(size_t)r * HDIM + c]; s += v; q += v * v; }
  atomicAdd(&sum[c], s); atomicAdd(&sq[c], q);
}
__global__ void bn_fin_kernel(const float* sum, const float* sq, const float* __restrict__ gamma,
                              float* mu, float* a, int N) {
  int c = threadIdx.x;
  if (c < HDIM) {
    float m   = sum[c] / (float)N;
    float var = sq[c] / (float)N - m * m;         // biased var (jnp.var)
    mu[c] = m;
    a[c]  = gamma[c] * rsqrtf(var + 1e-5f);
  }
}
__global__ void bn_apply_kernel(float* __restrict__ h, const float* __restrict__ hn,
                                const float* __restrict__ mu, const float* __restrict__ a,
                                const float* __restrict__ beta, _Float16* __restrict__ h16, int N) {
  int i = blockIdx.x * blockDim.x + threadIdx.x;
  if (i >= N * HDIM) return;
  int c = i & 127;
  float v  = (hn[i] - mu[c]) * a[c] + beta[c];
  float hv = h[i] + fmaxf(v, 0.f);
  h[i] = hv; h16[i] = (_Float16)hv;
}

// ------------------------------------------------ row L2 normalize (wave per row, 64 cols)
__global__ void row_norm_kernel(float* __restrict__ o, int N) {
  int lane = threadIdx.x & 31;
  int n = blockIdx.x * 8 + (threadIdx.x >> 5);
  if (n >= N) return;
  float2 v = *(const float2*)(o + (size_t)n * ODIM + lane * 2);
  float s = v.x * v.x + v.y * v.y;
#pragma unroll
  for (int off = 16; off; off >>= 1) s += __shfl_xor(s, off);
  float inv = 1.f / fmaxf(sqrtf(s), 1e-12f);
  o[(size_t)n * ODIM + lane * 2]     = v.x * inv;
  o[(size_t)n * ODIM + lane * 2 + 1] = v.y * inv;
}

// ================================================================ launch
extern "C" void kernel_launch(void* const* d_in, const int* in_sizes, int n_in,
                              void* d_out, int out_size, void* d_ws, size_t ws_size,
                              hipStream_t stream) {
  (void)n_in; (void)out_size; (void)ws_size;
  const int N  = in_sizes[0] / 6;
  const int E  = in_sizes[1] / 2;
  const int ET = E + N;

  const float* x     = (const float*)d_in[0];
  const int*   ei    = (const int*)d_in[1];
  const float* ipw   = (const float*)d_in[2];
  const float* ipb   = (const float*)d_in[3];
  const float* gatw  = (const float*)d_in[4];
  const float* asrc  = (const float*)d_in[5];
  const float* adst  = (const float*)d_in[6];
  const float* gatb  = (const float*)d_in[7]; (void)gatb; // cancelled exactly by BN mean
  const float* gamma = (const float*)d_in[8];
  const float* beta  = (const float*)d_in[9];
  const float* mhaw  = (const float*)d_in[10];
  const float* mhab  = (const float*)d_in[11];
  const float* mow   = (const float*)d_in[12];
  const float* mob   = (const float*)d_in[13];
  const float* oww   = (const float*)d_in[14];
  const float* obb   = (const float*)d_in[15];
  float* out = (float*)d_out;

  uintptr_t base = (uintptr_t)d_ws;
  auto alloc = [&](size_t bytes) -> void* {
    base = (base + 255) & ~(uintptr_t)255;
    void* p = (void*)base; base += bytes; return p;
  };
  float*    h    = (float*)   alloc((size_t)N * HDIM * 4);
  _Float16* h16  = (_Float16*)alloc((size_t)N * HDIM * 2);
  float*    xh   = (float*)   alloc((size_t)N * HDIM * 4);
  float*    hn   = (float*)   alloc((size_t)N * HDIM * 4);
  _Float16* v16w = (_Float16*)alloc((size_t)N * HDIM * 2);
  float*    es   = (float*)   alloc((size_t)N * HEADS * 4);
  float*    ed   = (float*)   alloc((size_t)N * HEADS * 4);
  unsigned* mk   = (unsigned*)alloc((size_t)N * HEADS * 4);
  float*    den  = (float*)   alloc((size_t)N * HEADS * 4);
  _Float16* gw16 = (_Float16*)alloc((size_t)3 * HDIM * HDIM * 2);
  _Float16* wv16 = (_Float16*)alloc((size_t)HDIM * HDIM * 2);
  _Float16* wo16 = (_Float16*)alloc((size_t)HDIM * HDIM * 2);
  _Float16* ow16 = (_Float16*)alloc((size_t)ODIM * HDIM * 2);
  float*    st   = (float*)   alloc(1024 * 4);
  float *sum4 = st, *sq4 = st + 4, *mean4 = st + 8, *inv4 = st + 12;
  float *bnsum = st + 16, *bnsq = st + 144, *bnmu = st + 272, *bna = st + 400;

  const int TPB = 256;
  auto cdiv = [](long long a, long long b) { return (int)((a + b - 1) / b); };

  // f16 weight staging
  f32_to_f16_kernel<<<cdiv(3 * HDIM * HDIM, TPB), TPB, 0, stream>>>(gatw, gw16, 3 * HDIM * HDIM);
  f32_to_f16_kernel<<<cdiv(HDIM * HDIM, TPB), TPB, 0, stream>>>(mhaw + 2 * HDIM * HDIM, wv16, HDIM * HDIM);
  f32_to_f16_kernel<<<cdiv(HDIM * HDIM, TPB), TPB, 0, stream>>>(mow, wo16, HDIM * HDIM);
  f32_to_f16_kernel<<<cdiv(ODIM * HDIM, TPB), TPB, 0, stream>>>(oww, ow16, ODIM * HDIM);

  // input normalization + projection
  fill_f32_kernel<<<1, 64, 0, stream>>>(st, 16, 0.f);
  col_stats4_kernel<<<cdiv(N, 1024), 128, 0, stream>>>(x, sum4, sq4, N);
  in_stats_fin_kernel<<<1, 32, 0, stream>>>(sum4, sq4, mean4, inv4, N);
  in_proj_kernel<<<cdiv((long long)N * HDIM, TPB), TPB, 0, stream>>>(x, ipw, ipb, mean4, inv4, h, h16, N);

  const int gemmBlk128 = cdiv((long long)(N / 16) * (HDIM / 64), 8);  // 16x64 strips
  for (int l = 0; l < 3; ++l) {
    wmma_gemm_kernel<false, false, true, false><<<gemmBlk128, 256, 0, stream>>>(
        h16, gw16 + (size_t)l * HDIM * HDIM, nullptr, nullptr, xh, nullptr, N, HDIM);
    att_scores_kernel<<<cdiv((long long)N * HEADS, TPB), TPB, 0, stream>>>(
        xh, asrc + l * HEADS * DHD, adst + l * HEADS * DHD, es, ed, N);
    fill_u32_kernel<<<cdiv((long long)N * HEADS, TPB), TPB, 0, stream>>>(mk, N * HEADS, 0u);
    fill_f32_kernel<<<cdiv((long long)N * HEADS, TPB), TPB, 0, stream>>>(den, N * HEADS, 0.f);
    fill_f32_kernel<<<cdiv((long long)N * HDIM, TPB), TPB, 0, stream>>>(hn, N * HDIM, 0.f);
    edge_max_kernel<<<cdiv((long long)ET * HEADS, TPB), TPB, 0, stream>>>(ei, E, ET, es, ed, mk);
    edge_sum_kernel<<<cdiv((long long)ET * HEADS, TPB), TPB, 0, stream>>>(ei, E, ET, es, ed, mk, den);
    edge_agg_kernel<<<cdiv((long long)ET * 32, TPB), TPB, 0, stream>>>(ei, E, ET, es, ed, mk, den, xh, hn);
    fill_f32_kernel<<<1, 256, 0, stream>>>(bnsum, 256, 0.f);   // zeros bnsum + bnsq
    bn_stats_kernel<<<cdiv(N, 256), 128, 0, stream>>>(hn, bnsum, bnsq, N);
    bn_fin_kernel<<<1, 128, 0, stream>>>(bnsum, bnsq, gamma + l * HDIM, bnmu, bna, N);
    bn_apply_kernel<<<cdiv((long long)N * HDIM, TPB), TPB, 0, stream>>>(
        h, hn, bnmu, bna, beta + l * HDIM, h16, N);
  }

  // MHA value path: v = h@Wv^T + bv ; h += v@Wo^T + bo
  wmma_gemm_kernel<true, false, false, true><<<gemmBlk128, 256, 0, stream>>>(
      h16, wv16, mhab + 2 * HDIM, nullptr, nullptr, v16w, N, HDIM);
  wmma_gemm_kernel<true, true, true, true><<<gemmBlk128, 256, 0, stream>>>(
      v16w, wo16, mob, h, h, h16, N, HDIM);

  // output projection + row normalize
  const int gemmBlk64 = cdiv((long long)(N / 16) * (ODIM / 64), 8);
  wmma_gemm_kernel<true, false, true, false><<<gemmBlk64, 256, 0, stream>>>(
      h16, ow16, obb, nullptr, out, nullptr, N, ODIM);
  row_norm_kernel<<<cdiv(N, 8), 256, 0, stream>>>(out, N);
}
